// LSTMPosPennModel_20830591386078
// MI455X (gfx1250) — matrix operand order repaired and checked by reading, not compiled, not run on hardware
//
#include <hip/hip_runtime.h>
#include <hip/hip_bf16.h>
#include <stdint.h>

// ---------------------------------------------------------------------------
// Problem constants (match reference)
// ---------------------------------------------------------------------------
#define N_TAGS 36
#define E_DIM  300
#define E_PAD  320      // padded K for bf16 WMMA (multiple of 32)
#define H_DIM  512
#define B_SZ   64
#define T_LEN  200
#define C_OUT  5
#define G4     2048     // 4*H (gates i,f,g,o)

typedef __attribute__((ext_vector_type(16))) __bf16 v16bf;
typedef __attribute__((ext_vector_type(8)))  __bf16 v8bf;
typedef __attribute__((ext_vector_type(8)))  float  v8f;

__device__ __forceinline__ v16bf joinbf(v8bf lo, v8bf hi) {
    return __builtin_shufflevector(lo, hi, 0,1,2,3,4,5,6,7,8,9,10,11,12,13,14,15);
}
__device__ __forceinline__ float sigm(float x) { return 1.0f / (1.0f + __expf(-x)); }

// ---------------------------------------------------------------------------
// Phase 0: init state (h double-buffer, grid-barrier counters)
// ---------------------------------------------------------------------------
__global__ __launch_bounds__(256) void k_init(__bf16* __restrict__ hbuf,
                                              unsigned* __restrict__ bar) {
    int idx = blockIdx.x * 256 + threadIdx.x;
    if (idx < 2 * B_SZ * H_DIM) hbuf[idx] = (__bf16)0.0f;
    if (idx < 2) bar[idx] = 0u;
}

// ---------------------------------------------------------------------------
// Phase 1a: weight conversion fp32 -> bf16 (W_ih padded 300->320 with zeros)
// ---------------------------------------------------------------------------
__global__ __launch_bounds__(256) void k_cvt_wih(const float* __restrict__ w,
                                                 __bf16* __restrict__ o) {
    long idx = (long)blockIdx.x * 256 + threadIdx.x;
    const long total = (long)N_TAGS * G4 * E_PAD;
    if (idx >= total) return;
    int  c   = (int)(idx % E_PAD);
    long row = idx / E_PAD;
    o[idx] = (c < E_DIM) ? (__bf16)w[row * E_DIM + c] : (__bf16)0.0f;
}

__global__ __launch_bounds__(256) void k_cvt_whh(const float* __restrict__ w,
                                                 __bf16* __restrict__ o) {
    long idx = (long)blockIdx.x * 256 + threadIdx.x;
    const long total = (long)N_TAGS * G4 * H_DIM;
    if (idx >= total) return;
    o[idx] = (__bf16)w[idx];
}

// ---------------------------------------------------------------------------
// Phase 1b: embedding gather -> xbf[t][b][E_PAD] (bf16, zero padded)
// ---------------------------------------------------------------------------
__global__ __launch_bounds__(256) void k_embed(const int* __restrict__ tokens,
                                               const float* __restrict__ emb,
                                               __bf16* __restrict__ xbf) {
    long idx = (long)blockIdx.x * 256 + threadIdx.x;   // over T*B*(E_PAD/2)
    const long total = (long)T_LEN * B_SZ * (E_PAD / 2);
    if (idx >= total) return;
    int  c2 = (int)(idx % (E_PAD / 2)) * 2;
    long tb = idx / (E_PAD / 2);
    int  b  = (int)(tb % B_SZ);
    int  t  = (int)(tb / B_SZ);
    int  tok = tokens[b * T_LEN + t];
    const float* er = emb + (long)tok * E_DIM;
    float f0 = (c2     < E_DIM) ? er[c2]     : 0.0f;
    float f1 = (c2 + 1 < E_DIM) ? er[c2 + 1] : 0.0f;
    __bf16* p = xbf + tb * E_PAD + c2;
    p[0] = (__bf16)f0;
    p[1] = (__bf16)f1;
}

// ---------------------------------------------------------------------------
// Phase 1c (parallel over all T): Xg[t] = x_t @ W_ih[tag_t]^T + b_ih + b_hh
// One wave per 16x16 output tile; K = 320 (10 WMMA per tile).
// ---------------------------------------------------------------------------
__global__ __launch_bounds__(256, 2) void k_xg_gemm(const __bf16* __restrict__ xbf,
                                                    const __bf16* __restrict__ wih,
                                                    const float*  __restrict__ bih,
                                                    const float*  __restrict__ bhh,
                                                    const int*    __restrict__ tags,
                                                    float*        __restrict__ Xg) {
    int lane = threadIdx.x & 31;
    int tile = blockIdx.x * 8 + (threadIdx.x >> 5);  // 8 waves / block
    int t    = tile >> 9;          // 512 tiles per timestep (4 row * 128 col)
    int rem  = tile & 511;
    int r    = rem >> 7;           // batch row block (0..3)
    int cblk = rem & 127;          // gate column block (0..127)
    int tag  = tags[t];
    int nL   = lane & 15;
    int g    = lane >> 4;
    int col  = cblk * 16 + nL;

    float bv = bih[tag * G4 + col] + bhh[tag * G4 + col];
    v8f acc;
#pragma unroll
    for (int i = 0; i < 8; ++i) acc[i] = bv;

    const __bf16* xa = xbf + ((long)t * B_SZ + r * 16 + nL) * E_PAD;      // A row (M = lane&15)
    const __bf16* wb = wih + ((long)tag * G4 + col) * E_PAD;              // B row (N = lane&15)

#pragma unroll
    for (int k0 = 0; k0 < E_PAD; k0 += 32) {
        v8bf a0 = *(const v8bf*)(xa + k0 + 8 * g);
        v8bf a1 = *(const v8bf*)(xa + k0 + 16 + 8 * g);
        v8bf b0 = *(const v8bf*)(wb + k0 + 8 * g);
        v8bf b1 = *(const v8bf*)(wb + k0 + 16 + 8 * g);
        acc = __builtin_amdgcn_wmma_f32_16x16x32_bf16(false, joinbf(a0, a1),
                                                      false, joinbf(b0, b1),
                                                      (short)0, acc, false, false);
    }

    float* out = Xg + (long)t * B_SZ * G4;
#pragma unroll
    for (int rr = 0; rr < 8; ++rr)
        out[(r * 16 + rr + 8 * g) * G4 + col] = acc[rr];
}

// ---------------------------------------------------------------------------
// Phase 2: persistent recurrent scan. 32 workgroups (one per 16-hidden-unit
// block) x 512 threads (16 waves: 4 batch-row blocks x 4 gates). Per step:
// gate tile = Xg tile + h @ W_hh[tag]^T (16 WMMA), LDS gate exchange,
// pointwise LSTM cell update, device-wide sense barrier.
// ---------------------------------------------------------------------------
__global__ __launch_bounds__(512, 1) void k_scan(const int*    __restrict__ tags,
                                                 const float*  __restrict__ Xg,
                                                 const __bf16* __restrict__ whh,
                                                 __bf16*       __restrict__ hbuf,
                                                 float*        __restrict__ hF,
                                                 unsigned*     __restrict__ bar) {
    __shared__ float gbuf[4][B_SZ * 16];   // i/f/g/o pre-activations for this block
    __shared__ float cbuf[B_SZ * 16];      // resident cell state

    const int cb   = blockIdx.x;           // hidden-unit block 0..31
    const int tid  = threadIdx.x;
    const int lane = tid & 31;
    const int wave = tid >> 5;
    const int r    = wave & 3;             // batch row block
    const int q    = wave >> 2;            // gate index (i,f,g,o)
    const int nL   = lane & 15;
    const int g    = lane >> 4;
    const int colG = q * 512 + cb * 16 + nL;   // column in 4H gate space

    cbuf[tid]       = 0.0f;
    cbuf[tid + 512] = 0.0f;
    __syncthreads();

    unsigned* barCount = bar;
    unsigned* barGen   = bar + 1;

    for (int t = 0; t < T_LEN; ++t) {
        const int tag = tags[t];
        const __bf16* hread  = hbuf + (t & 1) * (B_SZ * H_DIM);
        __bf16*       hwrite = hbuf + ((t & 1) ^ 1) * (B_SZ * H_DIM);

        // C init: precomputed input-gate contribution (+ both biases)
        const float* xg = Xg + (long)t * B_SZ * G4;
        v8f acc;
#pragma unroll
        for (int rr = 0; rr < 8; ++rr)
            acc[rr] = xg[(r * 16 + rr + 8 * g) * G4 + colG];

        // h @ W_hh[tag]^T over K=512
        const __bf16* ha = hread + (r * 16 + nL) * H_DIM;
        const __bf16* wb = whh + ((long)tag * G4 + q * 512 + cb * 16 + nL) * H_DIM;
#pragma unroll
        for (int k0 = 0; k0 < H_DIM; k0 += 32) {
            v8bf a0 = *(const v8bf*)(ha + k0 + 8 * g);
            v8bf a1 = *(const v8bf*)(ha + k0 + 16 + 8 * g);
            v8bf b0 = *(const v8bf*)(wb + k0 + 8 * g);
            v8bf b1 = *(const v8bf*)(wb + k0 + 16 + 8 * g);
            acc = __builtin_amdgcn_wmma_f32_16x16x32_bf16(false, joinbf(a0, a1),
                                                          false, joinbf(b0, b1),
                                                          (short)0, acc, false, false);
        }

        // exchange gate tiles through LDS
#pragma unroll
        for (int rr = 0; rr < 8; ++rr)
            gbuf[q][(r * 16 + rr + 8 * g) * 16 + nL] = acc[rr];
        __syncthreads();

        // pointwise LSTM cell update; each thread owns 2 cells (64x16 = 1024)
        for (int cidx = tid; cidx < B_SZ * 16; cidx += 512) {
            float ii = gbuf[0][cidx];
            float ff = gbuf[1][cidx];
            float gg = gbuf[2][cidx];
            float oo = gbuf[3][cidx];
            float c  = cbuf[cidx];
            c = sigm(ff) * c + sigm(ii) * tanhf(gg);
            float h = sigm(oo) * tanhf(c);
            cbuf[cidx] = c;
            int m  = cidx >> 4;
            int nn = cidx & 15;
            hwrite[m * H_DIM + cb * 16 + nn] = (__bf16)h;
            if (t == T_LEN - 1) hF[m * H_DIM + cb * 16 + nn] = h;
        }

        // ---- device-wide sense barrier (h_write must be visible to all) ----
        __builtin_amdgcn_fence(__ATOMIC_RELEASE, "agent");
        __syncthreads();
        if (tid == 0) {
            unsigned sense = __hip_atomic_load(barGen, __ATOMIC_RELAXED,
                                               __HIP_MEMORY_SCOPE_AGENT);
            unsigned ticket = __hip_atomic_fetch_add(barCount, 1u, __ATOMIC_ACQ_REL,
                                                     __HIP_MEMORY_SCOPE_AGENT);
            if (ticket == gridDim.x - 1) {
                __hip_atomic_store(barCount, 0u, __ATOMIC_RELAXED,
                                   __HIP_MEMORY_SCOPE_AGENT);
                __hip_atomic_fetch_add(barGen, 1u, __ATOMIC_RELEASE,
                                       __HIP_MEMORY_SCOPE_AGENT);
            } else {
                while (__hip_atomic_load(barGen, __ATOMIC_ACQUIRE,
                                         __HIP_MEMORY_SCOPE_AGENT) == sense) {
                    __builtin_amdgcn_s_sleep(2);
                }
            }
        }
        __syncthreads();
        __builtin_amdgcn_fence(__ATOMIC_ACQUIRE, "agent");  // invalidate L0 for fresh h
    }
}

// ---------------------------------------------------------------------------
// Phase 3: dense head (tiny; scalar VALU is fine)
// ---------------------------------------------------------------------------
__global__ __launch_bounds__(256) void k_head1(const float* __restrict__ hF,
                                               const float* __restrict__ W1,
                                               const float* __restrict__ b1,
                                               float* __restrict__ hid) {
    int b = blockIdx.x, j = threadIdx.x;
    const float* hv = hF + b * H_DIM;
    const float* wv = W1 + j * H_DIM;
    float s = 0.0f;
#pragma unroll 8
    for (int k = 0; k < H_DIM; ++k) s += hv[k] * wv[k];
    s += b1[j];
    hid[b * 256 + j] = s > 0.0f ? s : 0.0f;
}

__global__ __launch_bounds__(320) void k_head2(const float* __restrict__ hid,
                                               const float* __restrict__ W2,
                                               const float* __restrict__ b2,
                                               float* __restrict__ out) {
    int idx = threadIdx.x;
    if (idx >= B_SZ * C_OUT) return;
    int b = idx / C_OUT, c = idx % C_OUT;
    float s = 0.0f;
#pragma unroll 8
    for (int k = 0; k < 256; ++k) s += hid[b * 256 + k] * W2[c * 256 + k];
    out[idx] = s + b2[c];
}

// ---------------------------------------------------------------------------
extern "C" void kernel_launch(void* const* d_in, const int* in_sizes, int n_in,
                              void* d_out, int out_size, void* d_ws, size_t ws_size,
                              hipStream_t stream) {
    (void)in_sizes; (void)n_in; (void)out_size; (void)ws_size;

    const int*   tokens = (const int*)d_in[0];
    const int*   tags   = (const int*)d_in[1];
    const float* emb    = (const float*)d_in[2];
    const float* W_ih   = (const float*)d_in[3];
    const float* W_hh   = (const float*)d_in[4];
    const float* b_ih   = (const float*)d_in[5];
    const float* b_hh   = (const float*)d_in[6];
    const float* W1     = (const float*)d_in[7];
    const float* b1     = (const float*)d_in[8];
    const float* W2     = (const float*)d_in[9];
    const float* b2     = (const float*)d_in[10];
    float* out = (float*)d_out;

    char*  ws  = (char*)d_ws;
    size_t off = 0;
    auto alloc = [&](size_t bytes) -> char* {
        char* p = ws + off;
        off = (off + bytes + 255) & ~(size_t)255;
        return p;
    };
    __bf16*   xbf  = (__bf16*)alloc((size_t)T_LEN * B_SZ * E_PAD * 2);   //  8.2 MB
    __bf16*   wihb = (__bf16*)alloc((size_t)N_TAGS * G4 * E_PAD * 2);    // 47.2 MB
    __bf16*   whhb = (__bf16*)alloc((size_t)N_TAGS * G4 * H_DIM * 2);    // 75.5 MB
    float*    Xg   = (float*)alloc((size_t)T_LEN * B_SZ * G4 * 4);       // 104.9 MB
    __bf16*   hbuf = (__bf16*)alloc((size_t)2 * B_SZ * H_DIM * 2);
    float*    hF   = (float*)alloc((size_t)B_SZ * H_DIM * 4);
    float*    hid  = (float*)alloc((size_t)B_SZ * 256 * 4);
    unsigned* bar  = (unsigned*)alloc(256);

    // Phase 0/1: init + convert + gather (all parallel)
    k_init<<<256, 256, 0, stream>>>(hbuf, bar);
    {
        long tot = (long)N_TAGS * G4 * E_PAD;
        k_cvt_wih<<<(unsigned)((tot + 255) / 256), 256, 0, stream>>>(W_ih, wihb);
    }
    {
        long tot = (long)N_TAGS * G4 * H_DIM;
        k_cvt_whh<<<(unsigned)((tot + 255) / 256), 256, 0, stream>>>(W_hh, whhb);
    }
    {
        long tot = (long)T_LEN * B_SZ * (E_PAD / 2);
        k_embed<<<(unsigned)((tot + 255) / 256), 256, 0, stream>>>(tokens, emb, xbf);
    }

    // Phase 1c: input-gate GEMM for all timesteps (WMMA)
    {
        unsigned tiles = T_LEN * 512;          // 102400 tiles, 8 waves/block
        k_xg_gemm<<<tiles / 8, 256, 0, stream>>>(xbf, wihb, b_ih, b_hh, tags, Xg);
    }

    // Phase 2: persistent recurrent scan (WMMA + grid barrier)
    k_scan<<<32, 512, 0, stream>>>(tags, Xg, whhb, hbuf, hF, bar);

    // Phase 3: head
    k_head1<<<B_SZ, 256, 0, stream>>>(hF, W1, b1, hid);
    k_head2<<<1, 320, 0, stream>>>(hid, W2, b2, out);
}
